// STFT_14946486190601
// MI455X (gfx1250) — matrix operand order repaired and checked by reading, not compiled, not run on hardware
//
#include <hip/hip_runtime.h>

typedef __attribute__((ext_vector_type(16))) _Float16 v16h;
typedef __attribute__((ext_vector_type(8)))  _Float16 v8h;
typedef __attribute__((ext_vector_type(8)))  float    v8f;
typedef __attribute__((ext_vector_type(2)))  float    v2f;

#define N_FFT     1024
#define HOP       256
#define BATCH     32
#define SAMPLES   524288
#define FRAMES    2048        // (SAMPLES + 768 - N_FFT)/HOP + 1
#define BINS      513         // N_FFT/2 + 1
#define BIN_TILES 33          // ceil(513/16)
#define ROWS_PAD  528         // 33*16 padded rows of the DFT matrix
#define PAD_LEFT  384         // (N_FFT - HOP)/2
#define XP_LEN    (SAMPLES + 768)   // 525056 padded sample count
#define XP_STRIDE 525312            // XP_LEN rounded up (mult of 256, 32B align)

// ---------------------------------------------------------------------------
// Setup 1: bake window into f16 DFT matrices.
//   Wc[k][n] =  window[n] * cos(2*pi*k*n/1024)
//   Ws[k][n] = -window[n] * sin(2*pi*k*n/1024)
// Angle reduced exactly via (k*n) mod 1024 (cos/sin are 1024-periodic in k*n).
// ---------------------------------------------------------------------------
__global__ void build_dft_f16(const float* __restrict__ window,
                              _Float16* __restrict__ Wc,
                              _Float16* __restrict__ Ws) {
  int idx = blockIdx.x * blockDim.x + threadIdx.x;
  if (idx >= ROWS_PAD * N_FFT) return;
  int k = idx >> 10;
  int n = idx & 1023;
  int t = (k * n) & 1023;                       // exact reduction mod N_FFT
  float phi = (float)t * (6.283185307179586f / 1024.0f);
  float c = __cosf(phi);
  float s = __sinf(phi);
  float w = window[n];
  Wc[idx] = (_Float16)(w * c);
  Ws[idx] = (_Float16)(-w * s);
}

// ---------------------------------------------------------------------------
// Setup 2: convert x to f16 with the reflect-free zero padding baked in.
//   xp[b][i] = (0 <= i-384 < SAMPLES) ? (f16)x[b][i-384] : 0
// After this, the STFT inner loop needs no bounds checks and no converts.
// ---------------------------------------------------------------------------
__global__ void pad_convert_f16(const float* __restrict__ x,
                                _Float16* __restrict__ xp) {
  int i = blockIdx.x * blockDim.x + threadIdx.x;
  if (i >= XP_STRIDE) return;
  int b = blockIdx.y;
  long s = (long)i - PAD_LEFT;
  float v = (s >= 0 && s < (long)SAMPLES) ? x[(size_t)b * SAMPLES + s] : 0.0f;
  xp[(size_t)b * XP_STRIDE + i] = (_Float16)v;
}

// ---------------------------------------------------------------------------
// WMMA helper
// ---------------------------------------------------------------------------
static __device__ __forceinline__ v8f wmma_f16(v16h a, v16h b, v8f c) {
  return __builtin_amdgcn_wmma_f32_16x16x32_f16(false, a, false, b,
                                                (short)0, c, false, false);
}

// ---------------------------------------------------------------------------
// Main STFT kernel: one wave computes 16 bins x 64 frames (complex).
//   D[k,f] = sum_n W[k,n] * xp[b, f*HOP + n]
// K-loop in chunks of 32: 4 frame tiles x (re,im) = 8 WMMAs per chunk,
// A fragments loaded once per chunk and reused by all 4 frame tiles.
// All loads are unconditional contiguous 16B f16 vector loads.
// ---------------------------------------------------------------------------
__global__ void __launch_bounds__(256)
stft_wmma_kernel(const _Float16* __restrict__ xp,
                 const _Float16* __restrict__ Wc,
                 const _Float16* __restrict__ Ws,
                 float*          __restrict__ out) {
  const int lane    = threadIdx.x & 31;
  const int wave    = threadIdx.x >> 5;
  const int binTile = blockIdx.y * 8 + wave;     // uniform per wave
  if (binTile >= BIN_TILES) return;              // whole-wave exit only

  const int  ftg = blockIdx.x;                   // 0..1023 frame groups
  const int  b   = ftg >> 5;                     // batch (32 groups per batch)
  const int  F0  = (ftg & 31) << 6;              // first frame of 64-frame group
  const int  f0  = F0 + (lane & 15);             // lane's frame, tile 0
  const bool hi  = lane >= 16;
  const int  rowA = binTile * 16 + (lane & 15);  // lane's DFT row (A matrix)

  const int offA = hi ? 16 : 0;                  // A K-offset (contiguous k = 16h + s)
  const int offB = hi ? 16 : 0;                  // B K-offset (contiguous k = 16h + s)

  const _Float16* rowC = Wc + (size_t)rowA * N_FFT;
  const _Float16* rowS = Ws + (size_t)rowA * N_FFT;
  // Base of this lane's B column (tile 0) in the padded f16 signal.
  const _Float16* xb0 = xp + (size_t)b * XP_STRIDE + (size_t)f0 * HOP + offB;

  v8f cRe0 = {}, cIm0 = {}, cRe1 = {}, cIm1 = {};
  v8f cRe2 = {}, cIm2 = {}, cRe3 = {}, cIm3 = {};

  for (int k0 = 0; k0 < N_FFT; k0 += 32) {
    // A fragments: halves 0..7 = K k0+offA.., halves 8..15 = +8 (2x16B each)
    union { v16h v; v8h h[2]; } aRe, aIm;
    aRe.h[0] = *(const v8h*)(rowC + k0 + offA);
    aRe.h[1] = *(const v8h*)(rowC + k0 + offA + 8);
    aIm.h[0] = *(const v8h*)(rowS + k0 + offA);
    aIm.h[1] = *(const v8h*)(rowS + k0 + offA + 8);

    // B fragments: 16 consecutive padded f16 samples per lane per tile.
    union { v16h v; v8h h[2]; } b0, b1, b2, b3;
    b0.h[0] = *(const v8h*)(xb0 + k0);
    b0.h[1] = *(const v8h*)(xb0 + k0 + 8);
    b1.h[0] = *(const v8h*)(xb0 + 16 * HOP + k0);
    b1.h[1] = *(const v8h*)(xb0 + 16 * HOP + k0 + 8);
    b2.h[0] = *(const v8h*)(xb0 + 32 * HOP + k0);
    b2.h[1] = *(const v8h*)(xb0 + 32 * HOP + k0 + 8);
    b3.h[0] = *(const v8h*)(xb0 + 48 * HOP + k0);
    b3.h[1] = *(const v8h*)(xb0 + 48 * HOP + k0 + 8);

    cRe0 = wmma_f16(aRe.v, b0.v, cRe0);
    cIm0 = wmma_f16(aIm.v, b0.v, cIm0);
    cRe1 = wmma_f16(aRe.v, b1.v, cRe1);
    cIm1 = wmma_f16(aIm.v, b1.v, cIm1);
    cRe2 = wmma_f16(aRe.v, b2.v, cRe2);
    cIm2 = wmma_f16(aIm.v, b2.v, cIm2);
    cRe3 = wmma_f16(aRe.v, b3.v, cRe3);
    cIm3 = wmma_f16(aIm.v, b3.v, cIm3);
  }

  // Store: C/D layout VGPR r -> bin = 16*binTile + r + (hi?8:0), col = frame.
  // Output [B, 513, 2048] complex64 -> fused (re,im) 8B non-temporal stores.
  const size_t rowBase = ((size_t)b * BINS + binTile * 16) * FRAMES;
#pragma unroll
  for (int r = 0; r < 8; ++r) {
    int mr = r + (hi ? 8 : 0);                   // row within tile
    if (binTile < BIN_TILES - 1 || binTile * 16 + mr < BINS) {
      size_t o = (rowBase + (size_t)mr * FRAMES + f0) * 2;
      v2f p0 = {cRe0[r], cIm0[r]};
      v2f p1 = {cRe1[r], cIm1[r]};
      v2f p2 = {cRe2[r], cIm2[r]};
      v2f p3 = {cRe3[r], cIm3[r]};
      __builtin_nontemporal_store(p0, (v2f*)(out + o));
      __builtin_nontemporal_store(p1, (v2f*)(out + o + 32));   // +16 frames
      __builtin_nontemporal_store(p2, (v2f*)(out + o + 64));   // +32 frames
      __builtin_nontemporal_store(p3, (v2f*)(out + o + 96));   // +48 frames
    }
  }
}

extern "C" void kernel_launch(void* const* d_in, const int* in_sizes, int n_in,
                              void* d_out, int out_size, void* d_ws, size_t ws_size,
                              hipStream_t stream) {
  const float* x      = (const float*)d_in[0];   // [32, 524288] f32
  const float* window = (const float*)d_in[1];   // [1024] f32
  float* out = (float*)d_out;                    // [32, 513, 2048] complex64

  // Workspace: Wc | Ws (528*1024 f16 each) | padded f16 signal (32*525312 f16)
  _Float16* Wc = (_Float16*)d_ws;
  _Float16* Ws = Wc + (size_t)ROWS_PAD * N_FFT;
  _Float16* xp = Ws + (size_t)ROWS_PAD * N_FFT;

  int totalW = ROWS_PAD * N_FFT;
  build_dft_f16<<<(totalW + 255) / 256, 256, 0, stream>>>(window, Wc, Ws);

  dim3 gpad((XP_STRIDE + 255) / 256, BATCH);
  pad_convert_f16<<<gpad, 256, 0, stream>>>(x, xp);

  // 1024 frame groups (32 batches * 32 groups of 64 frames),
  // 5 y-blocks * 8 waves >= 33 bin tiles.
  dim3 grid(BATCH * FRAMES / 64, (BIN_TILES + 7) / 8);
  stft_wmma_kernel<<<grid, 256, 0, stream>>>(xp, Wc, Ws, out);
}
